// MultiHeadAttention_5274219839561
// MI455X (gfx1250) — matrix operand order repaired and checked by reading, not compiled, not run on hardware
//
#include <hip/hip_runtime.h>
#include <hip/hip_bf16.h>

// ---------------------------------------------------------------------------
// MHA for MI455X (gfx1250): bf16 WMMA GEMMs + flash attention (online softmax)
// B=2, S=4096, D=512, H=8, d_k=64
// ---------------------------------------------------------------------------

typedef __bf16 v16bf __attribute__((ext_vector_type(16)));
typedef float  v8f   __attribute__((ext_vector_type(8)));
typedef unsigned int v4u __attribute__((ext_vector_type(4)));
typedef unsigned int v8u __attribute__((ext_vector_type(8)));

union FragA { v16bf v; uint4 u[2]; };

__device__ __forceinline__ v8f wmma_bf16(v16bf a, v16bf b, v8f c) {
  // D = A(16x32 bf16) * B(32x16 bf16) + C(16x16 f32)
  return __builtin_amdgcn_wmma_f32_16x16x32_bf16(
      /*neg_a=*/false, a, /*neg_b=*/false, b,
      /*c_mod=*/(short)0, c, /*reuse_a=*/false, /*reuse_b=*/false);
}

// 16-lane butterfly reductions via DPP16 (no LDS round-trips, no dscnt waits)
template <int CTRL>
__device__ __forceinline__ float dppmov(float x) {
  return __int_as_float(
      __builtin_amdgcn_update_dpp(0, __float_as_int(x), CTRL, 0xf, 0xf, true));
}
__device__ __forceinline__ float rowmax16(float x) {
  x = fmaxf(x, dppmov<0xB1>(x));    // quad_perm(1,0,3,2)  : xor 1
  x = fmaxf(x, dppmov<0x4E>(x));    // quad_perm(2,3,0,1)  : xor 2
  x = fmaxf(x, dppmov<0x141>(x));   // row_half_mirror
  x = fmaxf(x, dppmov<0x140>(x));   // row_mirror
  return x;
}
__device__ __forceinline__ float rowsum16(float x) {
  x += dppmov<0xB1>(x);
  x += dppmov<0x4E>(x);
  x += dppmov<0x141>(x);
  x += dppmov<0x140>(x);
  return x;
}

// ------------------------------ convert ------------------------------------
__global__ void cvt_f32_bf16(const float* __restrict__ in,
                             __bf16* __restrict__ out, int n) {
  int i = blockIdx.x * blockDim.x + threadIdx.x;
  int stride = gridDim.x * blockDim.x;
  for (; i < n; i += stride) out[i] = (__bf16)in[i];
}

// ------------------------------ GEMM ---------------------------------------
// Y[M,512] = A[M,512](bf16) @ W[512,512](bf16, row n = output col)^T + bias
// OUT_MODE 0: store bf16 head-split [B=2, H=8, S=4096, 64]
// OUT_MODE 1: store fp32 row-major [M, 512]
template <int OUT_MODE>
__global__ void __launch_bounds__(256, 2)
gemm_xwT(const __bf16* __restrict__ A,
         const __bf16* __restrict__ W,
         const float* __restrict__ bias,
         void* __restrict__ out) {
  const int K = 512;
  const int lane = threadIdx.x & 31;
  const int wave = threadIdx.x >> 5;
  const int half = lane >> 4;   // 0: lanes 0-15, 1: lanes 16-31
  const int l16  = lane & 15;
  const int m0 = blockIdx.x * 128 + wave * 16;
  const int n0 = blockIdx.y * 64;

  v8f acc[4] = {v8f{0}, v8f{0}, v8f{0}, v8f{0}};
  const __bf16* Arow = A + (size_t)(m0 + l16) * K;
  const int akoff = half * 16;

  for (int k = 0; k < K; k += 32) {
    FragA a;
    a.u[0] = *(const uint4*)(Arow + k + akoff);        // K chunk 0
    a.u[1] = *(const uint4*)(Arow + k + akoff + 8);    // K chunk 1
#pragma unroll
    for (int nt = 0; nt < 4; ++nt) {
      const __bf16* Wrow = W + (size_t)(n0 + nt * 16 + l16) * K + k + half * 16;
      FragA b;
      b.u[0] = *(const uint4*)(Wrow);
      b.u[1] = *(const uint4*)(Wrow + 8);
      acc[nt] = wmma_bf16(a.v, b.v, acc[nt]);
    }
  }

#pragma unroll
  for (int nt = 0; nt < 4; ++nt) {
#pragma unroll
    for (int r = 0; r < 8; ++r) {
      const int m = m0 + r + half * 8;           // row (token)
      const int n = n0 + nt * 16 + l16;          // col (feature)
      const float v = acc[nt][r] + bias[n];
      if (OUT_MODE == 0) {
        const int b = m >> 12, s = m & 4095;     // /4096, %4096
        const int h = n >> 6, dk = n & 63;
        ((__bf16*)out)[(((size_t)(b * 8 + h) * 4096) + s) * 64 + dk] = (__bf16)v;
      } else {
        ((float*)out)[(size_t)m * 512 + n] = v;
      }
    }
  }
}

// --------------------------- flash attention -------------------------------
// grid: (S/128, B*H); block: 256 (8 waves); wave owns 16 query rows.
// K tiles arrive via the Tensor Data Mover (double-buffered); V tiles are
// cooperatively transposed into LDS (double-buffered).
__global__ void __launch_bounds__(256, 1)
flash_attn(const __bf16* __restrict__ Qp,
           const __bf16* __restrict__ Kp,
           const __bf16* __restrict__ Vp,
           __bf16* __restrict__ Ctx) {
  const int S = 4096, DK = 64, NT = S / 64;
  __shared__ __bf16 Ks[2][64 * 72];     // K tile, row-major [key][dk], pad 72
  __shared__ __bf16 Vs[2][64 * 72];     // V tile, transposed [dk][key]
  __shared__ __bf16 Ps[8 * 16 * 64];    // per-wave P tile (row-major)

  const int lane = threadIdx.x & 31;
  const int wave = threadIdx.x >> 5;
  const int half = lane >> 4;
  const int l16  = lane & 15;
  const int bh = blockIdx.y;                 // b*8 + h
  const int q0 = blockIdx.x * 128 + wave * 16;

  const size_t headOff = (size_t)bh * S * DK;
  const __bf16* Qh = Qp + headOff;
  const __bf16* Kh = Kp + headOff;
  const __bf16* Vh = Vp + headOff;

  // Q fragments (A-layout), kept in registers for the whole kernel
  FragA qf[2];
  {
    const __bf16* qrow = Qh + (size_t)(q0 + l16) * DK;
#pragma unroll
    for (int ks = 0; ks < 2; ++ks) {
      const int koff = ks * 32 + half * 16;
      qf[ks].u[0] = *(const uint4*)(qrow + koff);
      qf[ks].u[1] = *(const uint4*)(qrow + koff + 8);
    }
  }

  v8f O[4] = {v8f{0}, v8f{0}, v8f{0}, v8f{0}};
  float mrun[8], lrun[8];
#pragma unroll
  for (int r = 0; r < 8; ++r) { mrun[r] = -1e30f; lrun[r] = 0.f; }

  const int trow = threadIdx.x >> 2;          // 0..63 (key row)
  const int tcol = (threadIdx.x & 3) * 16;    // 0,16,32,48 (dk chunk)
  const float scl = 0.125f;                   // 1/sqrt(64)

  // TDM D# group 1 (tile-invariant): data_size=2B; pad_enable;
  // pad_interval=128B (one 64-elem row); pad_amount=16B -> LDS row stride
  // 144B = 72 bf16 (matches Ks layout); tensor 64x64, tile 64x64, stride 64.
  v8u g1;
  g1[0] = (1u << 16) | (1u << 20) | (4u << 22) | (3u << 25);
  g1[1] = (64u & 0xffffu) << 16;              // tensor_dim0 @ bits63:48
  g1[2] = ((64u & 0xffffu) << 16);            // dim0 hi=0 | tensor_dim1 lo=64
  g1[3] = (64u << 16);                        // dim1 hi=0 | tile_dim0=64
  g1[4] = 64u;                                // tile_dim1=64, tile_dim2=0
  g1[5] = 64u;                                // tensor_dim0_stride=64 (lo32)
  g1[6] = 0u;
  g1[7] = 0u;

  auto issueK = [&](int j, int buf) {         // wave 0 only
    const unsigned long long ga =
        (unsigned long long)(uintptr_t)(Kh + (size_t)(j * 64) * DK);
    v4u g0;
    g0[0] = 1u;                                // count=1 (user descriptor)
    g0[1] = (unsigned)(uintptr_t)&Ks[buf][0];  // lds_addr (bytes)
    g0[2] = (unsigned)ga;                      // global_addr[31:0]
    g0[3] = (unsigned)(ga >> 32) | (2u << 30); // global_addr[56:32] | type=2
    asm volatile("tensor_load_to_lds %0, %1" :: "s"(g0), "s"(g1) : "memory");
  };
  auto loadV = [&](int j, int buf) {           // all threads
    const __bf16* vg = Vh + (size_t)(j * 64 + trow) * DK + tcol;
    union { uint4 u[2]; __bf16 e[16]; } vv;
    vv.u[0] = *(const uint4*)vg;
    vv.u[1] = *(const uint4*)(vg + 8);
#pragma unroll
    for (int e = 0; e < 16; ++e) Vs[buf][(tcol + e) * 72 + trow] = vv.e[e];
    if (j + 1 < NT) __builtin_prefetch(vg + 64 * DK, 0, 0);
  };

  // ------------------------------ prologue --------------------------------
  if (wave == 0) issueK(0, 0);
  loadV(0, 0);
  if (wave == 0) __builtin_amdgcn_s_wait_tensorcnt(0);
  __syncthreads();

  for (int j = 0; j < NT; ++j) {
    const int cur = j & 1, nxt = cur ^ 1;
    if (j + 1 < NT) {                     // stage next tile (overlaps compute)
      if (wave == 0) issueK(j + 1, nxt);
      loadV(j + 1, nxt);
    }

    // -------- scores: Q(16x64) @ K^T(64x64) -> 4 accumulators -------------
    v8f sc[4] = {v8f{0}, v8f{0}, v8f{0}, v8f{0}};
#pragma unroll
    for (int ks = 0; ks < 2; ++ks) {
#pragma unroll
      for (int nt = 0; nt < 4; ++nt) {
        const __bf16* kr = &Ks[cur][(nt * 16 + l16) * 72 + ks * 32 + half * 16];
        FragA kf;
        kf.u[0] = *(const uint4*)kr;
        kf.u[1] = *(const uint4*)(kr + 8);
        sc[nt] = wmma_bf16(qf[ks].v, kf.v, sc[nt]);
      }
    }

    // -------- online softmax (row = (vgpr r, lane-half)), DPP reductions --
#pragma unroll
    for (int r = 0; r < 8; ++r) {
      float mx = mrun[r];
#pragma unroll
      for (int nt = 0; nt < 4; ++nt) {
        float s = sc[nt][r] * scl;
        sc[nt][r] = s;
        mx = fmaxf(mx, s);
      }
      mx = rowmax16(mx);
      const float a = __expf(mrun[r] - mx);
      float rs = 0.f;
#pragma unroll
      for (int nt = 0; nt < 4; ++nt) {
        float p = __expf(sc[nt][r] - mx);
        sc[nt][r] = p;
        rs += p;
      }
      rs = rowsum16(rs);
      mrun[r] = mx;
      lrun[r] = lrun[r] * a + rs;
#pragma unroll
      for (int nt = 0; nt < 4; ++nt) O[nt][r] *= a;
    }

    // -------- P (D-layout) -> per-wave LDS -> A-layout fragments ----------
    __bf16* Pw = &Ps[wave * 16 * 64];
#pragma unroll
    for (int nt = 0; nt < 4; ++nt)
#pragma unroll
      for (int r = 0; r < 8; ++r)
        Pw[(r + half * 8) * 64 + nt * 16 + l16] = (__bf16)sc[nt][r];
    asm volatile("s_wait_dscnt 0" ::: "memory");  // wave-private region

    // -------- O += P(16x64) @ V(64x64) ------------------------------------
#pragma unroll
    for (int ks = 0; ks < 2; ++ks) {
      const __bf16* pr = Pw + l16 * 64 + ks * 32 + half * 16;
      FragA pf;
      pf.u[0] = *(const uint4*)pr;
      pf.u[1] = *(const uint4*)(pr + 8);
#pragma unroll
      for (int nt = 0; nt < 4; ++nt) {
        const __bf16* vr = &Vs[cur][(nt * 16 + l16) * 72 + ks * 32 + half * 16];
        FragA vf;
        vf.u[0] = *(const uint4*)vr;
        vf.u[1] = *(const uint4*)(vr + 8);
        O[nt] = wmma_bf16(pf.v, vf.v, O[nt]);
      }
    }

    if (wave == 0 && j + 1 < NT) __builtin_amdgcn_s_wait_tensorcnt(0);
    __syncthreads();   // next buffer (K via TDM, V via DS) is now visible
  }

  // -------- finalize: O /= l, store context [b, s, h*64+dk] ---------------
  const int b = bh >> 3, h = bh & 7;
#pragma unroll
  for (int r = 0; r < 8; ++r) {
    const float inv = 1.f / lrun[r];
    const int q = q0 + r + half * 8;
#pragma unroll
    for (int nt = 0; nt < 4; ++nt) {
      const int dk = nt * 16 + l16;
      Ctx[((size_t)(b * 4096 + q)) * 512 + h * 64 + dk] = (__bf16)(O[nt][r] * inv);
    }
  }
}

// ------------------------------ launch -------------------------------------
extern "C" void kernel_launch(void* const* d_in, const int* in_sizes, int n_in,
                              void* d_out, int out_size, void* d_ws, size_t ws_size,
                              hipStream_t stream) {
  const float* q  = (const float*)d_in[0];
  const float* k  = (const float*)d_in[1];
  const float* v  = (const float*)d_in[2];
  const float* Wq = (const float*)d_in[3];
  const float* bq = (const float*)d_in[4];
  const float* Wk = (const float*)d_in[5];
  const float* bk = (const float*)d_in[6];
  const float* Wv = (const float*)d_in[7];
  const float* bv = (const float*)d_in[8];
  const float* Wo = (const float*)d_in[9];
  const float* bo = (const float*)d_in[10];

  const int M = 8192;          // B*S
  const int D = 512;
  const size_t nX = (size_t)M * D;    // 4 Mi elems
  const size_t nW = (size_t)D * D;    // 256 Ki elems

  char* ws = (char*)d_ws;
  size_t off = 0;
  auto carve = [&](size_t bytes) {
    void* p = ws + off;
    off = (off + bytes + 255) & ~(size_t)255;
    return p;
  };
  __bf16* Xq  = (__bf16*)carve(nX * 2);
  __bf16* Xk  = (__bf16*)carve(nX * 2);
  __bf16* Xv  = (__bf16*)carve(nX * 2);
  __bf16* Wqb = (__bf16*)carve(nW * 2);
  __bf16* Wkb = (__bf16*)carve(nW * 2);
  __bf16* Wvb = (__bf16*)carve(nW * 2);
  __bf16* Wob = (__bf16*)carve(nW * 2);
  __bf16* Qp  = (__bf16*)carve(nX * 2);   // [B,H,S,64]
  __bf16* Kp  = (__bf16*)carve(nX * 2);
  __bf16* Vp  = (__bf16*)carve(nX * 2);
  __bf16* Ctx = (__bf16*)carve(nX * 2);   // [B,S,512]

  cvt_f32_bf16<<<1024, 256, 0, stream>>>(q,  Xq,  (int)nX);
  cvt_f32_bf16<<<1024, 256, 0, stream>>>(k,  Xk,  (int)nX);
  cvt_f32_bf16<<<1024, 256, 0, stream>>>(v,  Xv,  (int)nX);
  cvt_f32_bf16<<<256,  256, 0, stream>>>(Wq, Wqb, (int)nW);
  cvt_f32_bf16<<<256,  256, 0, stream>>>(Wk, Wkb, (int)nW);
  cvt_f32_bf16<<<256,  256, 0, stream>>>(Wv, Wvb, (int)nW);
  cvt_f32_bf16<<<256,  256, 0, stream>>>(Wo, Wob, (int)nW);

  dim3 gg(M / 128, D / 64);   // (64, 8)
  gemm_xwT<0><<<gg, 256, 0, stream>>>(Xq, Wqb, bq, Qp);
  gemm_xwT<0><<<gg, 256, 0, stream>>>(Xk, Wkb, bk, Kp);
  gemm_xwT<0><<<gg, 256, 0, stream>>>(Xv, Wvb, bv, Vp);

  flash_attn<<<dim3(4096 / 128, 16), 256, 0, stream>>>(Qp, Kp, Vp, Ctx);

  gemm_xwT<1><<<gg, 256, 0, stream>>>(Ctx, Wob, bo, d_out);
}
